// CostvolumeAttention_66546223284231
// MI455X (gfx1250) — matrix operand order repaired and checked by reading, not compiled, not run on hardware
//
#include <hip/hip_runtime.h>
#include <hip/hip_bf16.h>
#include <math.h>

// ---------------------------------------------------------------------------
// CostvolumeAttention on MI455X (gfx1250):
//   - all matmuls via v_wmma_f32_16x16x32_f16 (f16 operands, f32 accumulate)
//   - P@V attention kernel stages its probability tile via the Tensor Data
//     Mover (tensor_load_to_lds + s_wait_tensorcnt) from an f16 prob buffer
//   - float4 (b128) staging loads everywhere alignment allows
// ---------------------------------------------------------------------------
typedef __attribute__((ext_vector_type(16))) _Float16 v16h;
typedef __attribute__((ext_vector_type(8)))  float    v8f;
typedef __attribute__((ext_vector_type(4)))  unsigned int u32x4;
typedef __attribute__((ext_vector_type(8)))  int          i32x8;
typedef __attribute__((ext_vector_type(4)))  int          i32x4;

#if defined(__HIP_DEVICE_COMPILE__) && __has_builtin(__builtin_amdgcn_tensor_load_to_lds) && __has_builtin(__builtin_amdgcn_s_wait_tensorcnt)
#define CVA_USE_TDM 1
#else
#define CVA_USE_TDM 0
#endif

namespace cva {
constexpr int Bsz = 16, T = 8, HS = 14, WS = 14, HW = 196;
constexpr int NH = 8, HD = 96, C = 768, N = T * HW;       // 1568
constexpr int M1 = Bsz * N;                               // 25088
constexpr int BHT = Bsz * NH * T;                         // 1024
constexpr int C3 = 3 * C;                                 // 2304

// workspace layout (float units)
constexpr size_t QKV_ELEMS = (size_t)M1 * C3;             // region0: qkv -> S -> Y
constexpr size_t PER       = (size_t)BHT * HD * HW;       // 19,267,584
constexpr size_t QP_OFF    = QKV_ELEMS;
constexpr size_t KP_OFF    = QP_OFF + PER;
constexpr size_t VR_OFF    = KP_OFF + PER;
constexpr size_t CORR_OFF  = VR_OFF + PER;
constexpr size_t CORR_EL   = (size_t)BHT * 49 * HW;
constexpr size_t MOT_OFF   = CORR_OFF + CORR_EL;
constexpr size_t APP_OFF   = MOT_OFF + PER;
constexpr size_t PH_OFF    = APP_OFF + PER;               // f16 probs: BHT*HW*HW halves
constexpr size_t PH_FLOATS = ((size_t)BHT * HW * HW) / 2; // 19,668,992
constexpr size_t WS_FLOATS = PH_OFF + PH_FLOATS;          // ~735 MB
}

// ---- WMMA fragment helpers (wave32: row/col = lane%16, k = 16*(lane/16) + i for both A and B) ----
__device__ __forceinline__ v16h frag_a(const _Float16* s, int mBase, int ldk) {
  int lane = threadIdx.x & 31;
  const _Float16* p = s + (size_t)(mBase + (lane & 15)) * ldk + 16 * (lane >> 4);
  v16h a;
#pragma unroll
  for (int i = 0; i < 16; i++) a[i] = p[i];
  return a;
}
__device__ __forceinline__ v16h frag_b(const _Float16* s, int nBase, int ldk) {
  int lane = threadIdx.x & 31;
  const _Float16* p = s + (size_t)(nBase + (lane & 15)) * ldk + 16 * (lane >> 4);
  v16h b;
#pragma unroll
  for (int i = 0; i < 16; i++) b[i] = p[i];
  return b;
}
__device__ __forceinline__ v8f wmma16(v16h a, v16h b, v8f c) {
  return __builtin_amdgcn_wmma_f32_16x16x32_f16(false, a, false, b, (short)0, c,
                                                false, false);
}

// ---------------------------------------------------------------------------
// Generic GEMM: C[M,Nn] = A[M,K] @ B[K,Nn] (+bias). 128x128 block, BK=32.
// Requires M%128==0, Nn%128==0, K%32==0, rows 16B-aligned (all true here).
// ---------------------------------------------------------------------------
template <bool HAS_BIAS>
__global__ void gemm_wmma_kernel(const float* __restrict__ A,
                                 const float* __restrict__ B,
                                 const float* __restrict__ bias,
                                 float* __restrict__ Cout, int M, int Nn, int K) {
  __shared__ _Float16 As[128 * 32];   // [m][k]
  __shared__ _Float16 Bs[128 * 32];   // transposed: [n][k]
  const int nTiles = Nn >> 7;
  const int mT = blockIdx.x / nTiles;
  const int nT = blockIdx.x % nTiles;
  const int t = threadIdx.x;
  const int wave = t >> 5;
  const int wm = wave >> 1;
  const int wn = wave & 1;

  v8f acc[2][4];
#pragma unroll
  for (int i = 0; i < 2; i++)
#pragma unroll
    for (int j = 0; j < 4; j++) acc[i][j] = v8f{};

  const float* Ag = A + (size_t)mT * 128 * K;
  const float* Bg = B + (size_t)nT * 128;

  for (int k0 = 0; k0 < K; k0 += 32) {
#pragma unroll
    for (int i = 0; i < 4; i++) {                // A tile 128x32, b128 loads
      int e4 = t + i * 256;                      // 1024 float4s
      int r = e4 >> 3, c4 = (e4 & 7) << 2;
      float4 va = *(const float4*)(Ag + (size_t)r * K + k0 + c4);
      _Float16* d = &As[r * 32 + c4];
      d[0] = (_Float16)va.x; d[1] = (_Float16)va.y;
      d[2] = (_Float16)va.z; d[3] = (_Float16)va.w;
    }
#pragma unroll
    for (int i = 0; i < 4; i++) {                // B tile 32x128 -> Bs[n][k]
      int e4 = t + i * 256;
      int kk = e4 >> 5, c4 = (e4 & 31) << 2;
      float4 vb = *(const float4*)(Bg + (size_t)(k0 + kk) * Nn + c4);
      Bs[(c4 + 0) * 32 + kk] = (_Float16)vb.x;
      Bs[(c4 + 1) * 32 + kk] = (_Float16)vb.y;
      Bs[(c4 + 2) * 32 + kk] = (_Float16)vb.z;
      Bs[(c4 + 3) * 32 + kk] = (_Float16)vb.w;
    }
    if (k0 + 32 < K) {                           // global_prefetch_b8 next tiles
      __builtin_prefetch(Ag + (size_t)(t >> 1) * K + k0 + 32, 0, 3);
      __builtin_prefetch(Bg + (size_t)(k0 + 32 + (t >> 5)) * Nn + (t & 31) * 4, 0, 3);
    }
    __syncthreads();

    v16h af[2], bf[4];
#pragma unroll
    for (int i = 0; i < 2; i++) af[i] = frag_a(As, wm * 32 + i * 16, 32);
#pragma unroll
    for (int j = 0; j < 4; j++) bf[j] = frag_b(Bs, wn * 64 + j * 16, 32);
#pragma unroll
    for (int i = 0; i < 2; i++)
#pragma unroll
      for (int j = 0; j < 4; j++) acc[i][j] = wmma16(af[i], bf[j], acc[i][j]);
    __syncthreads();
  }

  const int lane = t & 31;
  const int nLoc = lane & 15;
  const int mOff = 8 * (lane >> 4);
#pragma unroll
  for (int i = 0; i < 2; i++)
#pragma unroll
    for (int j = 0; j < 4; j++) {
      int gn = nT * 128 + wn * 64 + j * 16 + nLoc;
      float bv = HAS_BIAS ? bias[gn] : 0.0f;
      size_t base = (size_t)(mT * 128 + wm * 32 + i * 16 + mOff) * Nn + gn;
#pragma unroll
      for (int r = 0; r < 8; r++) Cout[base + (size_t)r * Nn] = acc[i][j][r] + bv;
    }
}

// ---------------------------------------------------------------------------
// prep: qkv -> qp (l2norm q), kp (temporal-shift + l2norm k), vr (BHT,HW,HD)
// ---------------------------------------------------------------------------
__global__ void prep_kernel(const float* __restrict__ qkv, float* __restrict__ qp,
                            float* __restrict__ kp, float* __restrict__ vr) {
  using namespace cva;
  int idx = blockIdx.x * blockDim.x + threadIdx.x;
  if (idx >= BHT * HW) return;
  int p = idx % HW, bht = idx / HW;
  int tt = bht & 7, h = (bht >> 3) & 7, b = bht >> 6;
  int tk = (tt < T - 1) ? tt + 1 : T - 1;
  size_t rq = ((size_t)b * N + tt * HW + p) * C3;
  size_t rk = ((size_t)b * N + tk * HW + p) * C3;
  int col = h * HD;
  float sq = 0.f, sk = 0.f;
  for (int d = 0; d < HD; d += 4) {
    float4 a = *(const float4*)(qkv + rq + col + d);
    float4 c = *(const float4*)(qkv + rk + C + col + d);
    sq += a.x * a.x + a.y * a.y + a.z * a.z + a.w * a.w;
    sk += c.x * c.x + c.y * c.y + c.z * c.z + c.w * c.w;
  }
  float iq = 1.0f / fmaxf(sqrtf(sq), 1e-12f);
  float ik = 1.0f / fmaxf(sqrtf(sk), 1e-12f);
  size_t ob = (size_t)bht * HD * HW;
  float* vo = vr + (size_t)bht * HW * HD + (size_t)p * HD;
  for (int d = 0; d < HD; d += 4) {
    float4 a = *(const float4*)(qkv + rq + col + d);
    float4 c = *(const float4*)(qkv + rk + C + col + d);
    float4 v = *(const float4*)(qkv + rq + 2 * C + col + d);
    qp[ob + (size_t)(d + 0) * HW + p] = a.x * iq;
    qp[ob + (size_t)(d + 1) * HW + p] = a.y * iq;
    qp[ob + (size_t)(d + 2) * HW + p] = a.z * iq;
    qp[ob + (size_t)(d + 3) * HW + p] = a.w * iq;
    kp[ob + (size_t)(d + 0) * HW + p] = c.x * ik;
    kp[ob + (size_t)(d + 1) * HW + p] = c.y * ik;
    kp[ob + (size_t)(d + 2) * HW + p] = c.z * ik;
    kp[ob + (size_t)(d + 3) * HW + p] = c.w * ik;
    *(float4*)(vo + d) = v;
  }
}

// ---------------------------------------------------------------------------
// cost volume: corr[bht][u][p] = sum_d qp[d,p]*kp[d,p_shift] (zero pad) + pos
// ---------------------------------------------------------------------------
__global__ void corr_kernel(const float* __restrict__ qp, const float* __restrict__ kp,
                            const float* __restrict__ pos, float* __restrict__ corr) {
  using namespace cva;
  int bht = blockIdx.x;
  const float* q = qp + (size_t)bht * HD * HW;
  const float* k = kp + (size_t)bht * HD * HW;
  float* out = corr + (size_t)bht * 49 * HW;
  for (int e = threadIdx.x; e < 49 * HW; e += blockDim.x) {
    int u = e / HW, p = e % HW;
    int y = p / WS, x = p % WS;
    int yy = y + u / 7 - 3, xx = x + u % 7 - 3;
    float s = 0.0f;
    if (yy >= 0 && yy < HS && xx >= 0 && xx < WS) {
      int p2 = yy * WS + xx;
      for (int d = 0; d < HD; d++) s += q[(size_t)d * HW + p] * k[(size_t)d * HW + p2];
    }
    out[e] = s + pos[e];
  }
}

// ---------------------------------------------------------------------------
// batched scores: S[bht][n][m] = sum_k A[bht][k][n] * B[bht][k][m]
// grid (BHT, 4); block: 64 rows x 208 cols, K chunked by 32.
// ---------------------------------------------------------------------------
__global__ void score_kernel(const float* __restrict__ Ag, const float* __restrict__ Bg,
                             float* __restrict__ S, int K) {
  using namespace cva;
  __shared__ _Float16 As[64 * 32];
  __shared__ _Float16 Bs[208 * 32];
  const int bht = blockIdx.x;
  const int row0 = blockIdx.y * 64;
  const float* Ab = Ag + (size_t)bht * K * HW;
  const float* Bb = Bg + (size_t)bht * K * HW;
  const int t = threadIdx.x;
  const int wave = t >> 5;
  const int rt = wave >> 1;
  const int cg = wave & 1;

  v8f acc[7];
#pragma unroll
  for (int j = 0; j < 7; j++) acc[j] = v8f{};

  const int nKC = (K + 31) / 32;
  for (int kc = 0; kc < nKC; kc++) {
    int kb = kc * 32;
#pragma unroll
    for (int i = 0; i < 8; i++) {
      int e = t + i * 256, c = e >> 6, r = e & 63;
      int gk = kb + c, gr = row0 + r;
      float v = (gk < K && gr < HW) ? Ab[(size_t)gk * HW + gr] : 0.0f;
      As[r * 32 + c] = (_Float16)v;
    }
    for (int e = t; e < 208 * 32; e += 256) {
      int c = e / 208, m = e % 208;
      int gk = kb + c;
      float v = (gk < K && m < HW) ? Bb[(size_t)gk * HW + m] : 0.0f;
      Bs[m * 32 + c] = (_Float16)v;
    }
    __syncthreads();
    v16h af = frag_a(As, rt * 16, 32);
#pragma unroll
    for (int j = 0; j < 7; j++) {
      int ct = cg * 7 + j;
      if (ct < 13) {
        v16h bf = frag_b(Bs, ct * 16, 32);
        acc[j] = wmma16(af, bf, acc[j]);
      }
    }
    __syncthreads();
  }

  const int lane = t & 31;
  const int nLoc = lane & 15, mOff = 8 * (lane >> 4);
#pragma unroll
  for (int j = 0; j < 7; j++) {
    int ct = cg * 7 + j;
    if (ct >= 13) continue;
    int m = ct * 16 + nLoc;
    if (m >= HW) continue;
#pragma unroll
    for (int r = 0; r < 8; r++) {
      int n = row0 + rt * 16 + mOff + r;
      if (n < HW) S[(size_t)bht * HW * HW + (size_t)n * HW + m] = acc[j][r];
    }
  }
}

// ---------------------------------------------------------------------------
// row softmax, f32 scores in -> f16 probabilities out (wave per row)
// ---------------------------------------------------------------------------
__global__ void softmax_kernel(const float* __restrict__ S, _Float16* __restrict__ Ph,
                               int rows, int n) {
  int row = blockIdx.x * (blockDim.x >> 5) + (threadIdx.x >> 5);
  if (row >= rows) return;
  int lane = threadIdx.x & 31;
  const float* p = S + (size_t)row * n;
  _Float16* o = Ph + (size_t)row * n;
  float mx = -1e30f;
  for (int i = lane; i < n; i += 32) mx = fmaxf(mx, p[i]);
#pragma unroll
  for (int off = 16; off; off >>= 1) mx = fmaxf(mx, __shfl_xor(mx, off, 32));
  float sum = 0.0f;
  for (int i = lane; i < n; i += 32) sum += __expf(p[i] - mx);
#pragma unroll
  for (int off = 16; off; off >>= 1) sum += __shfl_xor(sum, off, 32);
  float inv = 1.0f / sum;
  for (int i = lane; i < n; i += 32) o[i] = (_Float16)(__expf(p[i] - mx) * inv);
}

// ---------------------------------------------------------------------------
// batched P@V: O[bht][n][d] = sum_m P[n][m] * V[m][d];  P is f16 (196x196).
// P tile (208 rows x 32 k) is DMA'd into LDS by the Tensor Data Mover: the
// D# tensor_dim1 (196) zero-fills row-padding; k-padding is not zeroed (dim0 stays 196 from Pb+kb).
// ---------------------------------------------------------------------------
__global__ void attnv_kernel(const _Float16* __restrict__ P, const float* __restrict__ V,
                             float* __restrict__ O) {
  using namespace cva;
  __shared__ _Float16 As[208 * 32];   // P rows [r][k]
  __shared__ _Float16 Bs[96 * 32];    // V^T [d][k]
  const int bht = blockIdx.x;
  const _Float16* Pb = P + (size_t)bht * HW * HW;
  const float* Vb = V + (size_t)bht * HW * HD;
  const int t = threadIdx.x;
  const int wave = t >> 5;
  const int rt0 = wave, rt1 = wave + 8;

  v8f acc[2][6];
#pragma unroll
  for (int i = 0; i < 2; i++)
#pragma unroll
    for (int j = 0; j < 6; j++) acc[i][j] = v8f{};

  for (int kc = 0; kc < 7; kc++) {
    int kb = kc * 32;
#if CVA_USE_TDM
    if (wave == 0) {
      unsigned long long ga =
          (unsigned long long)(uintptr_t)Pb + (unsigned long long)kb * 2ull;
      unsigned lds_addr = (unsigned)(uintptr_t)(&As[0]);
      u32x4 g0;
      g0[0] = 1u;                                   // count=1, user mode
      g0[1] = lds_addr;                             // LDS byte address
      g0[2] = (unsigned)(ga & 0xFFFFFFFFull);       // global_addr[31:0]
      g0[3] = (unsigned)((ga >> 32) & 0x01FFFFFFull) | (2u << 30);  // + type=2
      i32x8 g1;
      g1[0] = 1 << 16;                              // data_size=2B, no multicast
      g1[1] = (HW & 0xFFFF) << 16;                  // tensor_dim0 = 196
      g1[2] = (HW & 0xFFFF) << 16;                  // dim0 hi=0 | tensor_dim1 lo
      g1[3] = (32 << 16);                           // dim1 hi=0 | tile_dim0 = 32
      g1[4] = 208;                                  // tile_dim1 = 208, tile_dim2=0
      g1[5] = HW;                                   // tensor_dim0_stride = 196
      g1[6] = 0;
      g1[7] = 0;
      i32x4 g2 = {0, 0, 0, 0};
      i32x4 g3 = {0, 0, 0, 0};
#if __clang_major__ >= 23
      i32x8 g4 = {0, 0, 0, 0, 0, 0, 0, 0};
      __builtin_amdgcn_tensor_load_to_lds(g0, g1, g2, g3, g4, 0);
#else
      __builtin_amdgcn_tensor_load_to_lds(g0, g1, g2, g3, 0);
#endif
      __builtin_amdgcn_s_wait_tensorcnt(0);
    }
#else
#pragma unroll
    for (int i = 0; i < 26; i++) {               // manual As staging fallback
      int e = t + i * 256, c = e & 31, r = e >> 5;
      _Float16 v = (r < HW && kb + c < HW) ? Pb[(size_t)r * HW + kb + c]
                                           : (_Float16)0.0f;
      As[r * 32 + c] = v;
    }
#endif
#pragma unroll
    for (int i = 0; i < 3; i++) {                // Bs: V^T 96x32, b128 loads
      int e4 = t + i * 256;                      // < 768
      int c = e4 / 24, d4 = (e4 % 24) * 4;
      float4 v4 = {0.0f, 0.0f, 0.0f, 0.0f};
      if (kb + c < HW) v4 = *(const float4*)(Vb + (size_t)(kb + c) * HD + d4);
      Bs[(d4 + 0) * 32 + c] = (_Float16)v4.x;
      Bs[(d4 + 1) * 32 + c] = (_Float16)v4.y;
      Bs[(d4 + 2) * 32 + c] = (_Float16)v4.z;
      Bs[(d4 + 3) * 32 + c] = (_Float16)v4.w;
    }
    __syncthreads();
    v16h bf[6];
#pragma unroll
    for (int j = 0; j < 6; j++) bf[j] = frag_b(Bs, j * 16, 32);
    v16h a0 = frag_a(As, rt0 * 16, 32);
#pragma unroll
    for (int j = 0; j < 6; j++) acc[0][j] = wmma16(a0, bf[j], acc[0][j]);
    if (rt1 < 13) {
      v16h a1 = frag_a(As, rt1 * 16, 32);
#pragma unroll
      for (int j = 0; j < 6; j++) acc[1][j] = wmma16(a1, bf[j], acc[1][j]);
    }
    __syncthreads();
  }

  const int lane = t & 31;
  const int nLoc = lane & 15, mOff = 8 * (lane >> 4);
#pragma unroll
  for (int ii = 0; ii < 2; ii++) {
    int rt = ii ? rt1 : rt0;
    if (rt >= 13) continue;
#pragma unroll
    for (int j = 0; j < 6; j++) {
      int d = j * 16 + nLoc;
#pragma unroll
      for (int r = 0; r < 8; r++) {
        int row = rt * 16 + mOff + r;
        if (row < HW) O[(size_t)bht * HW * HD + (size_t)row * HD + d] = acc[ii][j][r];
      }
    }
  }
}

// ---------------------------------------------------------------------------
// merge: Y[b, t*HW+p, h*HD+d] = 0.5*(app + mot)
// ---------------------------------------------------------------------------
__global__ void merge_kernel(const float* __restrict__ app, const float* __restrict__ mot,
                             float* __restrict__ Y) {
  using namespace cva;
  int idx = blockIdx.x * blockDim.x + threadIdx.x;
  if (idx >= BHT * HW * HD) return;
  int d = idx % HD;
  int r2 = idx / HD;
  int p = r2 % HW, bht = r2 / HW;
  int tt = bht & 7, h = (bht >> 3) & 7, b = bht >> 6;
  float v = 0.5f * (app[idx] + mot[idx]);
  Y[((size_t)b * N + tt * HW + p) * C + h * HD + d] = v;
}

// ---------------------------------------------------------------------------
extern "C" void kernel_launch(void* const* d_in, const int* in_sizes, int n_in,
                              void* d_out, int out_size, void* d_ws, size_t ws_size,
                              hipStream_t stream) {
  using namespace cva;
  if (ws_size < WS_FLOATS * sizeof(float)) return;

  const float* x      = (const float*)d_in[0];
  const float* w_qkv  = (const float*)d_in[1];
  const float* w_proj = (const float*)d_in[2];
  const float* b_proj = (const float*)d_in[3];
  const float* pos    = (const float*)d_in[4];
  float* out = (float*)d_out;

  float* ws   = (float*)d_ws;
  float* qkv  = ws;                 // region0: qkv -> S -> Y (disjoint lifetimes)
  float* S    = ws;
  float* Y    = ws;
  float* qp   = ws + QP_OFF;
  float* kp   = ws + KP_OFF;
  float* vr   = ws + VR_OFF;
  float* corr = ws + CORR_OFF;
  float* motx = ws + MOT_OFF;
  float* appx = ws + APP_OFF;
  _Float16* Ph = (_Float16*)(ws + PH_OFF);

  // 1) qkv = x @ w_qkv (25088 x 2304, K=768)
  gemm_wmma_kernel<false><<<dim3((M1 / 128) * (C3 / 128)), 256, 0, stream>>>(
      x, w_qkv, nullptr, qkv, M1, C3, C);

  // 2) qp/kp (normalized, k shifted), vr
  prep_kernel<<<dim3((BHT * HW + 255) / 256), 256, 0, stream>>>(qkv, qp, kp, vr);

  // 3) cost volume + pos embed
  corr_kernel<<<dim3(BHT), 256, 0, stream>>>(qp, kp, pos, corr);

  // 4) motion: S = C^T C (K=49) -> softmax(f16) -> motx = P @ V
  score_kernel<<<dim3(BHT, 4), 256, 0, stream>>>(corr, corr, S, 49);
  softmax_kernel<<<dim3(BHT * HW / 8), 256, 0, stream>>>(S, Ph, BHT * HW, HW);
  attnv_kernel<<<dim3(BHT), 256, 0, stream>>>(Ph, vr, motx);

  // 5) appearance: S = qp^T kp (K=96) -> softmax(f16) -> appx = P @ V
  score_kernel<<<dim3(BHT, 4), 256, 0, stream>>>(qp, kp, S, 96);
  softmax_kernel<<<dim3(BHT * HW / 8), 256, 0, stream>>>(S, Ph, BHT * HW, HW);
  attnv_kernel<<<dim3(BHT), 256, 0, stream>>>(Ph, vr, appx);

  // 6) merge halves then project
  merge_kernel<<<dim3((BHT * HW * HD + 255) / 256), 256, 0, stream>>>(appx, motx, Y);
  gemm_wmma_kernel<true><<<dim3((M1 / 128) * (C / 128)), 256, 0, stream>>>(
      Y, w_proj, b_proj, out, M1, C, C);
}
